// ActorNetSpiking_77678778515914
// MI455X (gfx1250) — compile-verified, matrix operand order checked by reading, not run on hardware
//
#include <hip/hip_runtime.h>
#include <stdint.h>

typedef _Float16 v16h __attribute__((ext_vector_type(16)));
typedef _Float16 v8h  __attribute__((ext_vector_type(8)));
typedef float    v8f  __attribute__((ext_vector_type(8)));
typedef int      v4i  __attribute__((ext_vector_type(4)));

// Pointer-to-v4i in global (AS1) / LDS (AS3) address spaces, matching the
// probe-revealed signature of __builtin_amdgcn_global_load_async_to_lds_b128.
typedef __attribute__((address_space(1))) v4i* gas_v4i_ptr;
typedef __attribute__((address_space(3))) v4i* las_v4i_ptr;

#define BDIM   4096
#define SDIM   512
#define TDIM   50
#define HDIM   1024
#define ADIM   2

#define NEURON_CDECAY 0.5f
#define NEURON_VDECAY 0.75f
#define NEURON_VTH    0.5f

// gfx1250 async global->LDS path (ASYNCcnt), with portable fallback.
#if defined(__gfx1250__) && __has_builtin(__builtin_amdgcn_global_load_async_to_lds_b128) && __has_builtin(__builtin_amdgcn_s_wait_asynccnt)
#define USE_ASYNC_LDS 1
#else
#define USE_ASYNC_LDS 0
#endif

// ---------------------------------------------------------------------------
// Transpose + f16 convert: x [B,S,T] f32 -> xT [T,B,S] f16.
// Per-thread contiguous 200B read; writes coalesced across threads per t.
// ---------------------------------------------------------------------------
__global__ __launch_bounds__(256) void snn_transpose_x(
    const float* __restrict__ x, _Float16* __restrict__ xT) {
  int ms = blockIdx.x * 256 + threadIdx.x;        // 0 .. B*S-1
  const float* src = x + (size_t)ms * TDIM;
  float buf[TDIM];
#pragma unroll
  for (int t = 0; t < TDIM; ++t) buf[t] = src[t];
#pragma unroll
  for (int t = 0; t < TDIM; ++t)
    xT[(size_t)t * (BDIM * SDIM) + ms] = (_Float16)buf[t];
}

// All three weight matrices f32 -> f16 in one launch.
__global__ __launch_bounds__(256) void snn_convert_weights(
    const float* __restrict__ W1, const float* __restrict__ W2,
    const float* __restrict__ W3, _Float16* __restrict__ W1h,
    _Float16* __restrict__ W2h, _Float16* __restrict__ W3h) {
  int i = blockIdx.x * 256 + threadIdx.x;
  if (i < HDIM * SDIM) W1h[i] = (_Float16)W1[i];
  if (i < HDIM * HDIM) {
    W2h[i] = (_Float16)W2[i];
    W3h[i] = (_Float16)W3[i];
  }
}

__global__ __launch_bounds__(256) void zero_mem(uint32_t* p, size_t n) {
  size_t i = (size_t)blockIdx.x * blockDim.x + threadIdx.x;
  size_t stride = (size_t)gridDim.x * blockDim.x;
  for (; i < n; i += stride) p[i] = 0u;
}

// ---------------------------------------------------------------------------
// Fused GEMM + LIF neuron update for layers 1-3 (M=4096, N=1024).
//   pre = act @ W^T + bias   (f16 in, f32 accumulate, v_wmma_f32_16x16x32_f16)
//   u   = u*0.5 + pre ; v = v*0.75*(1-s_old) + u ; s = (v > 0.5)
// u/v/s all kept in f16 so the whole recurrent state (72 MB) stays L2-resident
// on MI455X (192 MB L2) -> per-step HBM traffic is essentially just x_t.
//
// Block = 128 threads = 4 waves (wave32), 2x2 wave grid, 64x64 per wave.
// BM = BN = 128, BK = 32, LDS double-buffered, one barrier per K-step,
// next tile streamed with global_load_async_to_lds_b128 under the WMMAs.
// ---------------------------------------------------------------------------
template <int K>
__global__ __launch_bounds__(128) void snn_layer_gemm(
    const _Float16* __restrict__ A,    // [4096, K] activations (f16)
    const _Float16* __restrict__ Wh,   // [1024, K] weights (f16)
    const float*    __restrict__ bias, // [1024]
    _Float16*       __restrict__ u,    // [4096, 1024] f16 state
    _Float16*       __restrict__ v,    // [4096, 1024] f16 state
    _Float16*       __restrict__ s)    // [4096, 1024] spikes (read old / write new)
{
  constexpr int N    = HDIM;
  constexpr int LDA  = 40;             // 32 halves + 8 pad (80B rows, 16B aligned)
  constexpr int BUFH = 128 * LDA;      // halves per buffer
  __shared__ _Float16 sA[2 * BUFH];
  __shared__ _Float16 sB[2 * BUFH];

  const int tid   = threadIdx.x;       // 0..127
  const int lane  = tid & 31;
  const int wave  = tid >> 5;          // 0..3
  const int waveM = wave & 1;          // 2 waves along M
  const int waveN = wave >> 1;         // 2 waves along N
  const int lh    = lane >> 4;         // half-wave select
  const int l15   = lane & 15;

  const int blockM = blockIdx.x * 128;
  const int blockN = blockIdx.y * 128;

  v8f c[4][4] = {};

  // Stage one 128x32 A tile + 128x32 B tile into LDS buffer `buf`.
  // 1024 x 16B chunks total -> 8 per thread (4 A + 4 B).
  auto stage = [&](int k0, int buf) {
#pragma unroll
    for (int j = 0; j < 4; ++j) {
      int chunk = tid + j * 128;                 // 0..511
      int row   = chunk >> 2;                    // 0..127
      int col8  = (chunk & 3) << 3;              // 0,8,16,24 halves
      const _Float16* ga = A  + (size_t)(blockM + row) * K + k0 + col8;
      const _Float16* gb = Wh + (size_t)(blockN + row) * K + k0 + col8;
      _Float16* la = sA + buf * BUFH + row * LDA + col8;
      _Float16* lb = sB + buf * BUFH + row * LDA + col8;
#if USE_ASYNC_LDS
      __builtin_amdgcn_global_load_async_to_lds_b128(
          (gas_v4i_ptr)ga, (las_v4i_ptr)la, 0, 0);
      __builtin_amdgcn_global_load_async_to_lds_b128(
          (gas_v4i_ptr)gb, (las_v4i_ptr)lb, 0, 0);
#else
      *(v8h*)la = *(const v8h*)ga;
      *(v8h*)lb = *(const v8h*)gb;
#endif
    }
  };

  stage(0, 0);
#if USE_ASYNC_LDS
  __builtin_amdgcn_s_wait_asynccnt(0);
#endif
  __syncthreads();

  constexpr int NK = K / 32;
  for (int kt = 0; kt < NK; ++kt) {
    const int cur = kt & 1;
    if (kt + 1 < NK) stage((kt + 1) * 32, cur ^ 1);  // overlap with WMMAs below

    const _Float16* bufA = sA + cur * BUFH;
    const _Float16* bufB = sB + cur * BUFH;

    // A fragments held across the whole nt loop (ISA 16-bit A layout:
    // two 8-half chunks at K = 8*lh and K = 16 + 8*lh).
    v16h af[4];
#pragma unroll
    for (int mt = 0; mt < 4; ++mt) {
      int r  = waveM * 64 + mt * 16 + l15;
      v8h lo = *(const v8h*)(bufA + r * LDA + 8 * lh);
      v8h hi = *(const v8h*)(bufA + r * LDA + 16 + 8 * lh);
      af[mt] = __builtin_shufflevector(lo, hi, 0,1,2,3,4,5,6,7,8,9,10,11,12,13,14,15);
    }
#pragma unroll
    for (int nt = 0; nt < 4; ++nt) {
      int r  = waveN * 64 + nt * 16 + l15;
      v8h lo = *(const v8h*)(bufB + r * LDA + 8 * lh);
      v8h hi = *(const v8h*)(bufB + r * LDA + 16 + 8 * lh);
      v16h bf = __builtin_shufflevector(lo, hi, 0,1,2,3,4,5,6,7,8,9,10,11,12,13,14,15);
#pragma unroll
      for (int mt = 0; mt < 4; ++mt)
        c[mt][nt] = __builtin_amdgcn_wmma_f32_16x16x32_f16(
            false, af[mt], false, bf, (short)0, c[mt][nt], false, false);
    }

#if USE_ASYNC_LDS
    __builtin_amdgcn_s_wait_asynccnt(0);   // next buffer's fills done
#endif
    __syncthreads();
  }

  // ---- fused LIF epilogue (f16 state) ----
  // C/D layout: VGPR r, lane -> M = r + 8*(lane>>4), N = lane&15.
#pragma unroll
  for (int mt = 0; mt < 4; ++mt) {
#pragma unroll
    for (int nt = 0; nt < 4; ++nt) {
      int col    = blockN + waveN * 64 + nt * 16 + l15;
      float bcol = bias[col];
#pragma unroll
      for (int r = 0; r < 8; ++r) {
        int row = blockM + waveM * 64 + mt * 16 + lh * 8 + r;
        size_t idx = (size_t)row * N + col;
        float pre = c[mt][nt][r] + bcol;
        float un  = (float)u[idx] * NEURON_CDECAY + pre;
        float so  = (float)s[idx];
        float vn  = (float)v[idx] * NEURON_VDECAY * (1.0f - so) + un;
        float sn  = (vn > NEURON_VTH) ? 1.0f : 0.0f;
        u[idx] = (_Float16)un;
        v[idx] = (_Float16)vn;
        s[idx] = (_Float16)sn;
      }
    }
  }
}

// ---------------------------------------------------------------------------
// Layer 4 (N = 2): per-row dot products + LIF + spike accumulation.
// ---------------------------------------------------------------------------
__global__ __launch_bounds__(256) void snn_layer4(
    const _Float16* __restrict__ s3,  // [4096, 1024]
    const float*    __restrict__ W4,  // [2, 1024]
    const float*    __restrict__ b4,  // [2]
    float* __restrict__ u4, float* __restrict__ v4,
    float* __restrict__ s4, float* __restrict__ acc)
{
  int m = blockIdx.x * 256 + threadIdx.x;       // 0..4095
  const _Float16* row = s3 + (size_t)m * HDIM;
  float a0 = 0.0f, a1 = 0.0f;
  for (int k = 0; k < HDIM; k += 8) {
    v8h sv = *(const v8h*)(row + k);
#pragma unroll
    for (int j = 0; j < 8; ++j) {
      float f = (float)sv[j];
      a0 += f * W4[k + j];
      a1 += f * W4[HDIM + k + j];
    }
  }
#pragma unroll
  for (int j = 0; j < ADIM; ++j) {
    float pre = (j ? a1 : a0) + b4[j];
    size_t idx = (size_t)m * ADIM + j;
    float un = u4[idx] * NEURON_CDECAY + pre;
    float so = s4[idx];
    float vn = v4[idx] * NEURON_VDECAY * (1.0f - so) + un;
    float sn = (vn > NEURON_VTH) ? 1.0f : 0.0f;
    u4[idx] = un;
    v4[idx] = vn;
    s4[idx] = sn;
    acc[idx] += sn;
  }
}

__global__ __launch_bounds__(256) void snn_finalize(
    const float* __restrict__ acc, float* __restrict__ out) {
  int i = blockIdx.x * 256 + threadIdx.x;
  if (i < BDIM * ADIM) out[i] = acc[i] * (1.0f / (float)TDIM);
}

// ---------------------------------------------------------------------------
// Host launcher
// ---------------------------------------------------------------------------
extern "C" void kernel_launch(void* const* d_in, const int* in_sizes, int n_in,
                              void* d_out, int out_size, void* d_ws, size_t ws_size,
                              hipStream_t stream) {
  (void)in_sizes; (void)n_in; (void)out_size;

  const float* x  = (const float*)d_in[0];
  const float* W1 = (const float*)d_in[1];
  const float* b1 = (const float*)d_in[2];
  const float* W2 = (const float*)d_in[3];
  const float* b2 = (const float*)d_in[4];
  const float* W3 = (const float*)d_in[5];
  const float* b3 = (const float*)d_in[6];
  const float* W4 = (const float*)d_in[7];
  const float* b4 = (const float*)d_in[8];

  // ---- workspace layout (256B aligned chunks) ----
  char*  base = (char*)d_ws;
  size_t off  = 0;
  auto take = [&](size_t bytes) -> char* {
    char* p = base + off;
    off += (bytes + 255) & ~(size_t)255;
    return p;
  };
  _Float16* xT  = (_Float16*)take((size_t)TDIM * BDIM * SDIM * 2); // 200 MB
  _Float16* W1h = (_Float16*)take((size_t)HDIM * SDIM * 2);
  _Float16* W2h = (_Float16*)take((size_t)HDIM * HDIM * 2);
  _Float16* W3h = (_Float16*)take((size_t)HDIM * HDIM * 2);
  char* stateBase = base + off;
  _Float16* u1 = (_Float16*)take((size_t)BDIM * HDIM * 2);
  _Float16* v1 = (_Float16*)take((size_t)BDIM * HDIM * 2);
  _Float16* s1 = (_Float16*)take((size_t)BDIM * HDIM * 2);
  _Float16* u2 = (_Float16*)take((size_t)BDIM * HDIM * 2);
  _Float16* v2 = (_Float16*)take((size_t)BDIM * HDIM * 2);
  _Float16* s2 = (_Float16*)take((size_t)BDIM * HDIM * 2);
  _Float16* u3 = (_Float16*)take((size_t)BDIM * HDIM * 2);
  _Float16* v3 = (_Float16*)take((size_t)BDIM * HDIM * 2);
  _Float16* s3 = (_Float16*)take((size_t)BDIM * HDIM * 2);
  float*    u4 = (float*)   take((size_t)BDIM * ADIM * 4);
  float*    v4 = (float*)   take((size_t)BDIM * ADIM * 4);
  float*    s4 = (float*)   take((size_t)BDIM * ADIM * 4);
  float*    ac = (float*)   take((size_t)BDIM * ADIM * 4);
  size_t stateBytes = (size_t)(((char*)ac + (size_t)BDIM * ADIM * 4) - stateBase);
  if (off > ws_size) return;  // workspace too small: do nothing (deterministic)

  // ---- per-call prep (deterministic) ----
  snn_transpose_x<<<(BDIM * SDIM) / 256, 256, 0, stream>>>(x, xT);
  snn_convert_weights<<<(HDIM * HDIM) / 256, 256, 0, stream>>>(W1, W2, W3, W1h, W2h, W3h);
  zero_mem<<<2048, 256, 0, stream>>>((uint32_t*)stateBase, stateBytes / 4);

  // ---- 50 timesteps, 4 fused layer kernels each (stream-ordered) ----
  dim3 gemmGrid(BDIM / 128, HDIM / 128);  // 32 x 8 = 256 blocks
  for (int t = 0; t < TDIM; ++t) {
    const _Float16* xt = xT + (size_t)t * BDIM * SDIM;
    snn_layer_gemm<SDIM><<<gemmGrid, 128, 0, stream>>>(xt, W1h, b1, u1, v1, s1);
    snn_layer_gemm<HDIM><<<gemmGrid, 128, 0, stream>>>(s1, W2h, b2, u2, v2, s2);
    snn_layer_gemm<HDIM><<<gemmGrid, 128, 0, stream>>>(s2, W3h, b3, u3, v3, s3);
    snn_layer4<<<BDIM / 256, 256, 0, stream>>>(s3, W4, b4, u4, v4, s4, ac);
  }
  snn_finalize<<<(BDIM * ADIM + 255) / 256, 256, 0, stream>>>(ac, (float*)d_out);
}
